// LightGlue_62560493633846
// MI455X (gfx1250) — matrix operand order repaired and checked by reading, not compiled, not run on hardware
//
#include <hip/hip_runtime.h>
#include <math.h>

#define B_ 2
#define N_ 1024
#define D_ 256
#define H_ 4
#define DH_ 64
#define L_ 6
#define M_ (B_*N_)
#define D2_ (2*D_)

typedef __bf16 bf16;
typedef __bf16 bf16x16 __attribute__((ext_vector_type(16)));
typedef __bf16 bf16x8  __attribute__((ext_vector_type(8)));
typedef float  f32x8   __attribute__((ext_vector_type(8)));
typedef float  f32x4   __attribute__((ext_vector_type(4)));

// ---------------------------------------------------------------- WMMA core
__device__ __forceinline__ f32x8 wmma_bf16(bf16x16 a, bf16x16 b, f32x8 c) {
  return __builtin_amdgcn_wmma_f32_16x16x32_bf16(false, a, false, b, (short)0, c,
                                                 false, false);
}

// A fragment (16x32 bf16): lane<16 holds row m=lane, k = kb..kb+7 and kb+16..kb+23
// (kb = kk); lanes 16..31 same rows with kb += 8.  Source is fp32, optionally
// split [A0 | A1] along K (for concat(x, m) @ W1).
__device__ __forceinline__ bf16x16 frag_a_f32(const float* A0, int K0,
                                              const float* A1, int K1,
                                              int row, int kk, int lane) {
  int kbase = kk + ((lane >> 4) << 3);
  bf16x16 a;
#pragma unroll
  for (int c = 0; c < 2; ++c) {
    int k0 = kbase + c * 16;
    const float* src = (k0 < K0) ? (A0 + (long)row * K0 + k0)
                                 : (A1 + (long)row * K1 + (k0 - K0));
    f32x4 lo = *(const f32x4*)(src);
    f32x4 hi = *(const f32x4*)(src + 4);
#pragma unroll
    for (int j = 0; j < 4; ++j) {
      a[c * 8 + j]     = (bf16)lo[j];
      a[c * 8 + 4 + j] = (bf16)hi[j];
    }
  }
  return a;
}

__device__ __forceinline__ bf16x16 frag_a_bf16(const bf16* A, int lda,
                                               int row, int kk, int lane) {
  int kbase = kk + ((lane >> 4) << 3);
  const bf16* p = A + (long)row * lda + kbase;
  bf16x8 lo = *(const bf16x8*)(p);
  bf16x8 hi = *(const bf16x8*)(p + 16);
  bf16x16 a;
#pragma unroll
  for (int j = 0; j < 8; ++j) { a[j] = lo[j]; a[8 + j] = hi[j]; }
  return a;
}

// B fragment (32x16 bf16): lane n=lane&15, k = kb..kb+7 and kb+16..kb+23,
// kb = kk + (lane>=16)*8.  Bt stored row-major [Nc x K] (i.e. B transposed).
__device__ __forceinline__ bf16x16 frag_b_bf16(const bf16* Bt, int ldb,
                                               int col, int kk, int lane) {
  int kb = kk + ((lane >> 4) << 3);
  const bf16* p = Bt + (long)col * ldb + kb;
  bf16x8 lo = *(const bf16x8*)(p);
  bf16x8 hi = *(const bf16x8*)(p + 16);
  bf16x16 b;
#pragma unroll
  for (int j = 0; j < 8; ++j) { b[j] = lo[j]; b[8 + j] = hi[j]; }
  return b;
}

// ------------------------------------------------- GEMM: fp32 A x bf16 Wt^T
// Cout[M x Nc] = alpha * (concat(A0,A1) @ Wt^T + bias) + residual
// 8 waves/block; wave tile = 16(M) x 64(N) (A fragment reused over 4 WMMAs);
// block tile = 64(M) x 128(N).  All Nc used here are multiples of 64.
__global__ void __launch_bounds__(256)
k_gemm(const float* A0, int K0, const float* A1, int K1,
       const bf16* __restrict__ Wt, const float* bias, const float* residual,
       float* Cout, bf16* Cbf, float alpha, int M, int Nc) {
  int lane  = threadIdx.x & 31;
  int w     = threadIdx.x >> 5;
  int waveN = blockIdx.x * 2 + (w & 1);   // 64-wide N strip
  int tileM = blockIdx.y * 4 + (w >> 1);
  if (tileM * 16 >= M || waveN * 64 >= Nc) return;
  int K    = K0 + K1;
  int rowA = tileM * 16 + (lane & 15);
  int colB = waveN * 64 + (lane & 15);
  f32x8 acc[4];
#pragma unroll
  for (int j = 0; j < 4; ++j)
#pragma unroll
    for (int i = 0; i < 8; ++i) acc[j][i] = 0.f;
  for (int kk = 0; kk < K; kk += 32) {
    if (kk + 32 < K)
      __builtin_prefetch(Wt + (long)colB * K + kk + 32, 0, 3);
    bf16x16 a = frag_a_f32(A0, K0, A1, K1, rowA, kk, lane);
#pragma unroll
    for (int j = 0; j < 4; ++j) {
      bf16x16 b = frag_b_bf16(Wt, K, colB + j * 16, kk, lane);
      acc[j] = wmma_bf16(a, b, acc[j]);
    }
  }
  int mb = tileM * 16 + ((lane >> 4) << 3);
#pragma unroll
  for (int j = 0; j < 4; ++j) {
    int   n  = waveN * 64 + j * 16 + (lane & 15);
    float bv = bias ? bias[n] : 0.f;
#pragma unroll
    for (int r = 0; r < 8; ++r) {
      long  o = (long)(mb + r) * Nc + n;
      float v = alpha * (acc[j][r] + bv);
      if (residual) v += residual[o];
      Cout[o] = v;
      if (Cbf) Cbf[o] = (bf16)v;
    }
  }
}

// --------------------------------------- GEMM NT: bf16 A [MxK] x bf16 Bt [NcxK]^T
// batched over blockIdx.z with explicit strides; f32 out with ldOut.
// Same 16x64 wave tile / 64x128 block tile.
__global__ void __launch_bounds__(256)
k_gemm_nt(const bf16* __restrict__ A, long sAb, const bf16* __restrict__ Bt, long sBb,
          float* Cout, long sCb, int ldOut, int M, int Nc, int K) {
  A    += (long)blockIdx.z * sAb;
  Bt   += (long)blockIdx.z * sBb;
  Cout += (long)blockIdx.z * sCb;
  int lane  = threadIdx.x & 31;
  int w     = threadIdx.x >> 5;
  int waveN = blockIdx.x * 2 + (w & 1);
  int tileM = blockIdx.y * 4 + (w >> 1);
  if (tileM * 16 >= M || waveN * 64 >= Nc) return;
  int rowA = tileM * 16 + (lane & 15);
  int colB = waveN * 64 + (lane & 15);
  f32x8 acc[4];
#pragma unroll
  for (int j = 0; j < 4; ++j)
#pragma unroll
    for (int i = 0; i < 8; ++i) acc[j][i] = 0.f;
  for (int kk = 0; kk < K; kk += 32) {
    bf16x16 a = frag_a_bf16(A, K, rowA, kk, lane);
#pragma unroll
    for (int j = 0; j < 4; ++j) {
      bf16x16 b = frag_b_bf16(Bt, K, colB + j * 16, kk, lane);
      acc[j] = wmma_bf16(a, b, acc[j]);
    }
  }
  int mb = tileM * 16 + ((lane >> 4) << 3);
#pragma unroll
  for (int j = 0; j < 4; ++j) {
    int n = waveN * 64 + j * 16 + (lane & 15);
#pragma unroll
    for (int r = 0; r < 8; ++r)
      Cout[(long)(mb + r) * ldOut + n] = acc[j][r];
  }
}

// ---------------------------------------------------------------- helpers
__global__ void k_convw(const float* __restrict__ W, bf16* __restrict__ Wt,
                        int K, int Nc) {
  int idx = blockIdx.x * blockDim.x + threadIdx.x;
  if (idx >= K * Nc) return;
  int k = idx / Nc, n = idx % Nc;
  Wt[(long)n * K + k] = (bf16)W[idx];
}

// rope/split-heads: fp32 [B,N,D] -> bf16 [B,H,N,DH]; use_rope applies axial RoPE
__global__ void k_rope(const float* __restrict__ qk, const float* __restrict__ kpts,
                       const float* __restrict__ fx, const float* __restrict__ fy,
                       bf16* __restrict__ out, int use_rope) {
  int idx = blockIdx.x * blockDim.x + threadIdx.x;
  if (idx >= B_ * N_ * D_) return;
  int d = idx & (DH_ - 1);
  int h = (idx >> 6) & (H_ - 1);
  int n = (idx >> 8) & (N_ - 1);
  int b = idx >> 18;
  float v = qk[idx];
  float r = v;
  if (use_rope) {
    float px  = (kpts[((long)b * N_ + n) * 2 + 0] - 512.f) * (1.f / 512.f);
    float py  = (kpts[((long)b * N_ + n) * 2 + 1] - 384.f) * (1.f / 512.f);
    float ang = (d < 32) ? px * fx[d >> 1] : py * fy[(d - 32) >> 1];
    float sn, cs;
    sincosf(ang, &sn, &cs);
    float p = (d & 1) ? qk[idx - 1] : -qk[idx + 1];  // rot_half pairing
    r = v * cs + p * sn;
  }
  out[(((long)b * H_ + h) * N_ + n) * DH_ + d] = (bf16)r;
}

// v fp32 [B,N,D] -> bf16 vT [B,H,DH,N]
__global__ void k_vt(const float* __restrict__ v, bf16* __restrict__ vT) {
  int idx = blockIdx.x * blockDim.x + threadIdx.x;
  if (idx >= B_ * N_ * D_) return;
  int d = idx & (DH_ - 1);
  int h = (idx >> 6) & (H_ - 1);
  int n = (idx >> 8) & (N_ - 1);
  int b = idx >> 18;
  vT[(((long)b * H_ + h) * DH_ + d) * N_ + n] = (bf16)v[idx];
}

__device__ __forceinline__ float warp_max(float v) {
#pragma unroll
  for (int o = 16; o > 0; o >>= 1) v = fmaxf(v, __shfl_xor(v, o, 32));
  return v;
}
__device__ __forceinline__ float warp_sum(float v) {
#pragma unroll
  for (int o = 16; o > 0; o >>= 1) v += __shfl_xor(v, o, 32);
  return v;
}
__device__ __forceinline__ float block_max(float v, float* sm) {
  v = warp_max(v);
  if ((threadIdx.x & 31) == 0) sm[threadIdx.x >> 5] = v;
  __syncthreads();
  float r = sm[0];
#pragma unroll
  for (int i = 1; i < 8; ++i) r = fmaxf(r, sm[i]);
  __syncthreads();
  return r;
}
__device__ __forceinline__ float block_sum(float v, float* sm) {
  v = warp_sum(v);
  if ((threadIdx.x & 31) == 0) sm[threadIdx.x >> 5] = v;
  __syncthreads();
  float r = sm[0];
#pragma unroll
  for (int i = 1; i < 8; ++i) r += sm[i];
  __syncthreads();
  return r;
}

// row softmax of per-head scores [B,N,N] -> bf16 probs (row-major)
__global__ void __launch_bounds__(256)
k_softmax_row(const float* __restrict__ S, bf16* __restrict__ P, float scale) {
  __shared__ float sm[8];
  int n = blockIdx.x, b = blockIdx.z;
  const float* row  = S + ((long)b * N_ + n) * N_;
  bf16*        prow = P + ((long)b * N_ + n) * N_;
  float vals[4];
  float mx = -3.402823466e38f;
#pragma unroll
  for (int t = 0; t < 4; ++t) {
    vals[t] = row[threadIdx.x + t * 256] * scale;
    mx = fmaxf(mx, vals[t]);
  }
  mx = block_max(mx, sm);
  float s = 0.f;
#pragma unroll
  for (int t = 0; t < 4; ++t) { vals[t] = __expf(vals[t] - mx); s += vals[t]; }
  s = block_sum(s, sm);
  float inv = 1.f / s;
#pragma unroll
  for (int t = 0; t < 4; ++t) prow[threadIdx.x + t * 256] = (bf16)(vals[t] * inv);
}

// column softmax (over n, per column m) -> transposed bf16 probs P[b][m][n]
__global__ void __launch_bounds__(256)
k_softmax_col(const float* __restrict__ S, bf16* __restrict__ P, float scale) {
  __shared__ float sm[8];
  int m = blockIdx.x, b = blockIdx.z;
  const float* base = S + (long)b * N_ * N_ + m;
  float vals[4];
  float mx = -3.402823466e38f;
#pragma unroll
  for (int t = 0; t < 4; ++t) {
    vals[t] = base[(long)(threadIdx.x + t * 256) * N_] * scale;
    mx = fmaxf(mx, vals[t]);
  }
  mx = block_max(mx, sm);
  float s = 0.f;
#pragma unroll
  for (int t = 0; t < 4; ++t) { vals[t] = __expf(vals[t] - mx); s += vals[t]; }
  s = block_sum(s, sm);
  float inv = 1.f / s;
  bf16* out = P + ((long)b * N_ + m) * N_;
#pragma unroll
  for (int t = 0; t < 4; ++t) out[threadIdx.x + t * 256] = (bf16)(vals[t] * inv);
}

__global__ void __launch_bounds__(256)
k_lse_row(const float* __restrict__ S, float* __restrict__ L) {
  __shared__ float sm[8];
  int n = blockIdx.x, b = blockIdx.z;
  const float* row = S + ((long)b * N_ + n) * N_;
  float vals[4];
  float mx = -3.402823466e38f;
#pragma unroll
  for (int t = 0; t < 4; ++t) { vals[t] = row[threadIdx.x + t * 256]; mx = fmaxf(mx, vals[t]); }
  mx = block_max(mx, sm);
  float s = 0.f;
#pragma unroll
  for (int t = 0; t < 4; ++t) s += expf(vals[t] - mx);
  s = block_sum(s, sm);
  if (threadIdx.x == 0) L[(long)b * N_ + n] = mx + logf(s);
}

__global__ void __launch_bounds__(256)
k_lse_col(const float* __restrict__ S, float* __restrict__ L) {
  __shared__ float sm[8];
  int m = blockIdx.x, b = blockIdx.z;
  const float* base = S + (long)b * N_ * N_ + m;
  float vals[4];
  float mx = -3.402823466e38f;
#pragma unroll
  for (int t = 0; t < 4; ++t) {
    vals[t] = base[(long)(threadIdx.x + t * 256) * N_];
    mx = fmaxf(mx, vals[t]);
  }
  mx = block_max(mx, sm);
  float s = 0.f;
#pragma unroll
  for (int t = 0; t < 4; ++t) s += expf(vals[t] - mx);
  s = block_sum(s, sm);
  if (threadIdx.x == 0) L[(long)b * N_ + m] = mx + logf(s);
}

// in-place LayerNorm + exact GELU over 512 features per row
__global__ void __launch_bounds__(256)
k_lngelu(float* __restrict__ Hb, const float* __restrict__ g,
         const float* __restrict__ be) {
  __shared__ float sm[8];
  long   row = blockIdx.x;
  float* p   = Hb + row * D2_;
  float v0 = p[threadIdx.x], v1 = p[threadIdx.x + 256];
  float mu = block_sum(v0 + v1, sm) * (1.f / D2_);
  float d0 = v0 - mu, d1 = v1 - mu;
  float var  = block_sum(d0 * d0 + d1 * d1, sm) * (1.f / D2_);
  float rstd = rsqrtf(var + 1e-5f);
  float y0 = d0 * rstd * g[threadIdx.x] + be[threadIdx.x];
  float y1 = d1 * rstd * g[threadIdx.x + 256] + be[threadIdx.x + 256];
  p[threadIdx.x]       = 0.5f * y0 * (1.f + erff(y0 * 0.70710678f));
  p[threadIdx.x + 256] = 0.5f * y1 * (1.f + erff(y1 * 0.70710678f));
}

// z[row] = x[row,:] . Wm + bm
__global__ void __launch_bounds__(256)
k_z(const float* __restrict__ x, const float* __restrict__ Wm,
    const float* __restrict__ bm, float* __restrict__ z) {
  __shared__ float sm[8];
  long  row = blockIdx.x;
  float v   = x[row * D_ + threadIdx.x] * Wm[threadIdx.x];
  float s   = block_sum(v, sm);
  if (threadIdx.x == 0) z[row] = s + bm[0];
}

__device__ __forceinline__ float logsig(float x) {
  return (x >= 0.f) ? -log1pf(expf(-x)) : (x - log1pf(expf(x)));
}

__global__ void k_assemble(const float* __restrict__ sim, const float* __restrict__ rl,
                           const float* __restrict__ cl, const float* __restrict__ z0,
                           const float* __restrict__ z1, float* __restrict__ out) {
  long idx = (long)blockIdx.x * blockDim.x + threadIdx.x;
  const long per = (long)(N_ + 1) * (N_ + 1);
  if (idx >= (long)B_ * per) return;
  int  b   = (int)(idx / per);
  long rem = idx - (long)b * per;
  int  r   = (int)(rem / (N_ + 1));
  int  c   = (int)(rem % (N_ + 1));
  float val;
  if (r < N_ && c < N_) {
    float s = sim[((long)b * N_ + r) * N_ + c];
    val = 2.f * s - rl[b * N_ + r] - cl[b * N_ + c]
        + logsig(z0[b * N_ + r]) + logsig(z1[b * N_ + c]);
  } else if (r < N_ && c == N_) {
    val = logsig(-z0[b * N_ + r]);
  } else if (r == N_ && c < N_) {
    val = logsig(-z1[b * N_ + c]);
  } else {
    val = 0.f;
  }
  out[idx] = val;
}

// ---------------------------------------------------------------- host
extern "C" void kernel_launch(void* const* d_in, const int* in_sizes, int n_in,
                              void* d_out, int out_size, void* d_ws, size_t ws_size,
                              hipStream_t stream) {
  (void)in_sizes; (void)n_in; (void)out_size; (void)ws_size;
  const float* kpts0 = (const float*)d_in[0];
  const float* kpts1 = (const float*)d_in[1];
  const float* desc0 = (const float*)d_in[2];
  const float* desc1 = (const float*)d_in[3];
  const float* fx    = (const float*)d_in[4];
  const float* fy    = (const float*)d_in[5];
  // per-set param order: Wqk,bqk,Wv,bv,Wo,bo,W1,b1,g,be,W2,b2 (set 0 = self, 1 = cross)
  const float* Wset[2][12];
  for (int p = 0; p < 2; ++p)
    for (int j = 0; j < 12; ++j) Wset[p][j] = (const float*)d_in[6 + p * 12 + j];
  const float* Wp = (const float*)d_in[30];
  const float* bp = (const float*)d_in[31];
  const float* Wm = (const float*)d_in[32];
  const float* bm = (const float*)d_in[33];

  char* cur = (char*)d_ws;
  auto alloc = [&](size_t bytes) -> void* {
    void* p = (void*)cur;
    cur += (bytes + 255) & ~(size_t)255;
    return p;
  };

  // ---- convert all weights to transposed bf16 (once per launch) ----
  bf16 *wqk[2][L_], *wv[2][L_], *wo[2][L_], *w1[2][L_], *w2[2][L_];
  auto convw = [&](const float* W, int K, int Nc) -> bf16* {
    bf16* t = (bf16*)alloc((size_t)K * Nc * sizeof(bf16));
    int tot = K * Nc;
    k_convw<<<(tot + 255) / 256, 256, 0, stream>>>(W, t, K, Nc);
    return t;
  };
  for (int p = 0; p < 2; ++p)
    for (int i = 0; i < L_; ++i) {
      wqk[p][i] = convw(Wset[p][0] + (long)i * D_ * D_, D_, D_);
      wv[p][i]  = convw(Wset[p][2] + (long)i * D_ * D_, D_, D_);
      wo[p][i]  = convw(Wset[p][4] + (long)i * D_ * D_, D_, D_);
      w1[p][i]  = convw(Wset[p][6] + (long)i * D2_ * D2_, D2_, D2_);
      w2[p][i]  = convw(Wset[p][10] + (long)i * D2_ * D_, D2_, D_);
    }
  bf16* wpT = convw(Wp, D_, D_);

  // ---- activation / scratch buffers ----
  float* x0   = (float*)alloc((size_t)M_ * D_ * 4);
  float* x1   = (float*)alloc((size_t)M_ * D_ * 4);
  float* Sc   = (float*)alloc((size_t)M_ * D2_ * 4);       // qk / h scratch
  bf16*  qr0  = (bf16*)alloc((size_t)B_ * H_ * N_ * DH_ * 2);
  bf16*  qr1  = (bf16*)alloc((size_t)B_ * H_ * N_ * DH_ * 2);
  bf16*  vT0  = (bf16*)alloc((size_t)B_ * H_ * DH_ * N_ * 2);
  bf16*  vT1  = (bf16*)alloc((size_t)B_ * H_ * DH_ * N_ * 2);
  float* AV0  = (float*)alloc((size_t)M_ * D_ * 4);
  float* AV1  = (float*)alloc((size_t)M_ * D_ * 4);
  float* MO0  = (float*)alloc((size_t)M_ * D_ * 4);
  float* MO1  = (float*)alloc((size_t)M_ * D_ * 4);
  float* simh = (float*)alloc((size_t)B_ * N_ * N_ * 4);   // per-head scores / final sim
  bf16*  Prow = (bf16*)alloc((size_t)B_ * N_ * N_ * 2);
  bf16*  Pcol = (bf16*)alloc((size_t)B_ * N_ * N_ * 2);
  float* z0   = (float*)alloc((size_t)M_ * 4);
  float* z1   = (float*)alloc((size_t)M_ * 4);
  float* rl   = (float*)alloc((size_t)M_ * 4);
  float* cl   = (float*)alloc((size_t)M_ * 4);

  hipMemcpyAsync(x0, desc0, (size_t)M_ * D_ * 4, hipMemcpyDeviceToDevice, stream);
  hipMemcpyAsync(x1, desc1, (size_t)M_ * D_ * 4, hipMemcpyDeviceToDevice, stream);

  auto gemm = [&](const float* A0, int K0, const float* A1, int K1, const bf16* Wt,
                  const float* bias, const float* res, float* Cout, bf16* Cbf,
                  float alpha, int M, int Nc) {
    dim3 g((Nc + 127) / 128, (M + 63) / 64, 1);
    k_gemm<<<g, 256, 0, stream>>>(A0, K0, A1, K1, Wt, bias, res, Cout, Cbf, alpha, M, Nc);
  };
  auto gemm_nt = [&](const bf16* A, long sA, const bf16* Bt, long sB,
                     float* C, long sC, int ldO, int M, int Nc, int K) {
    dim3 g((Nc + 127) / 128, (M + 63) / 64, B_);
    k_gemm_nt<<<g, 256, 0, stream>>>(A, sA, Bt, sB, C, sC, ldO, M, Nc, K);
  };

  const int  TOT_BND = B_ * N_ * D_;
  dim3 gbnd((TOT_BND + 255) / 256);
  const float SELF_SCALE = 0.125f;         // DH^-0.5
  const float CROSS_SC   = 0.35355339059f; // DH^-0.25 (applied to each qk)

  for (int i = 0; i < L_; ++i) {
    // ================= self blocks (shared-qk rotary attention) =================
    struct Str { float* x; const float* kp; bf16* qr; bf16* vt; float* av; float* mo; };
    Str st[2] = {{x0, kpts0, qr0, vT0, AV0, MO0}, {x1, kpts1, qr1, vT1, AV1, MO1}};
    const float* s_bqk = Wset[0][1] + i * D_;
    const float* s_bv  = Wset[0][3] + i * D_;
    const float* s_bo  = Wset[0][5] + i * D_;
    const float* s_b1  = Wset[0][7] + i * D2_;
    const float* s_g   = Wset[0][8] + i * D2_;
    const float* s_be  = Wset[0][9] + i * D2_;
    const float* s_b2  = Wset[0][11] + i * D_;
    for (int s = 0; s < 2; ++s) {
      gemm(st[s].x, D_, nullptr, 0, wqk[0][i], s_bqk, nullptr, Sc, nullptr, 1.f, M_, D_);
      k_rope<<<gbnd, 256, 0, stream>>>(Sc, st[s].kp, fx, fy, st[s].qr, 1);
      gemm(st[s].x, D_, nullptr, 0, wv[0][i], s_bv, nullptr, Sc, nullptr, 1.f, M_, D_);
      k_vt<<<gbnd, 256, 0, stream>>>(Sc, st[s].vt);
      for (int h = 0; h < H_; ++h) {
        gemm_nt(st[s].qr + (long)h * N_ * DH_, (long)H_ * N_ * DH_,
                st[s].qr + (long)h * N_ * DH_, (long)H_ * N_ * DH_,
                simh, (long)N_ * N_, N_, N_, N_, DH_);
        k_softmax_row<<<dim3(N_, 1, B_), 256, 0, stream>>>(simh, Prow, SELF_SCALE);
        gemm_nt(Prow, (long)N_ * N_,
                st[s].vt + (long)h * DH_ * N_, (long)H_ * DH_ * N_,
                st[s].av + h * DH_, (long)N_ * D_, D_, N_, DH_, N_);
      }
      gemm(st[s].av, D_, nullptr, 0, wo[0][i], s_bo, nullptr, st[s].mo, nullptr, 1.f, M_, D_);
      gemm(st[s].x, D_, st[s].mo, D_, w1[0][i], s_b1, nullptr, Sc, nullptr, 1.f, M_, D2_);
      k_lngelu<<<M_, 256, 0, stream>>>(Sc, s_g, s_be);
      gemm(Sc, D2_, nullptr, 0, w2[0][i], s_b2, st[s].x, st[s].x, nullptr, 1.f, M_, D_);
    }
    // ======================= bidirectional cross block =========================
    const float* c_bqk = Wset[1][1] + i * D_;
    const float* c_bv  = Wset[1][3] + i * D_;
    const float* c_bo  = Wset[1][5] + i * D_;
    const float* c_b1  = Wset[1][7] + i * D2_;
    const float* c_g   = Wset[1][8] + i * D2_;
    const float* c_be  = Wset[1][9] + i * D2_;
    const float* c_b2  = Wset[1][11] + i * D_;
    gemm(x0, D_, nullptr, 0, wqk[1][i], c_bqk, nullptr, Sc, nullptr, CROSS_SC, M_, D_);
    k_rope<<<gbnd, 256, 0, stream>>>(Sc, kpts0, fx, fy, qr0, 0);
    gemm(x1, D_, nullptr, 0, wqk[1][i], c_bqk, nullptr, Sc, nullptr, CROSS_SC, M_, D_);
    k_rope<<<gbnd, 256, 0, stream>>>(Sc, kpts1, fx, fy, qr1, 0);
    gemm(x0, D_, nullptr, 0, wv[1][i], c_bv, nullptr, Sc, nullptr, 1.f, M_, D_);
    k_vt<<<gbnd, 256, 0, stream>>>(Sc, vT0);
    gemm(x1, D_, nullptr, 0, wv[1][i], c_bv, nullptr, Sc, nullptr, 1.f, M_, D_);
    k_vt<<<gbnd, 256, 0, stream>>>(Sc, vT1);
    for (int h = 0; h < H_; ++h) {
      gemm_nt(qr0 + (long)h * N_ * DH_, (long)H_ * N_ * DH_,
              qr1 + (long)h * N_ * DH_, (long)H_ * N_ * DH_,
              simh, (long)N_ * N_, N_, N_, N_, DH_);
      k_softmax_row<<<dim3(N_, 1, B_), 256, 0, stream>>>(simh, Prow, 1.f);
      k_softmax_col<<<dim3(N_, 1, B_), 256, 0, stream>>>(simh, Pcol, 1.f);
      gemm_nt(Prow, (long)N_ * N_, vT1 + (long)h * DH_ * N_, (long)H_ * DH_ * N_,
              AV0 + h * DH_, (long)N_ * D_, D_, N_, DH_, N_);
      gemm_nt(Pcol, (long)N_ * N_, vT0 + (long)h * DH_ * N_, (long)H_ * DH_ * N_,
              AV1 + h * DH_, (long)N_ * D_, D_, N_, DH_, N_);
    }
    gemm(AV0, D_, nullptr, 0, wo[1][i], c_bo, nullptr, MO0, nullptr, 1.f, M_, D_);
    gemm(AV1, D_, nullptr, 0, wo[1][i], c_bo, nullptr, MO1, nullptr, 1.f, M_, D_);
    gemm(x0, D_, MO0, D_, w1[1][i], c_b1, nullptr, Sc, nullptr, 1.f, M_, D2_);
    k_lngelu<<<M_, 256, 0, stream>>>(Sc, c_g, c_be);
    gemm(Sc, D2_, nullptr, 0, w2[1][i], c_b2, x0, x0, nullptr, 1.f, M_, D_);
    gemm(x1, D_, MO1, D_, w1[1][i], c_b1, nullptr, Sc, nullptr, 1.f, M_, D2_);
    k_lngelu<<<M_, 256, 0, stream>>>(Sc, c_g, c_be);
    gemm(Sc, D2_, nullptr, 0, w2[1][i], c_b2, x1, x1, nullptr, 1.f, M_, D_);
  }

  // ======================= match assignment head =========================
  // md = (x @ Wp + bp) * D^-0.25 ; bf16 copies (reuse qr buffers) feed the NT GEMM
  gemm(x0, D_, nullptr, 0, wpT, bp, nullptr, Sc, qr0, 0.25f, M_, D_);
  gemm(x1, D_, nullptr, 0, wpT, bp, nullptr, Sc, qr1, 0.25f, M_, D_);
  gemm_nt(qr0, (long)N_ * D_, qr1, (long)N_ * D_, simh, (long)N_ * N_, N_, N_, N_, D_);
  k_z<<<M_, 256, 0, stream>>>(x0, Wm, bm, z0);
  k_z<<<M_, 256, 0, stream>>>(x1, Wm, bm, z1);
  k_lse_row<<<dim3(N_, 1, B_), 256, 0, stream>>>(simh, rl);
  k_lse_col<<<dim3(N_, 1, B_), 256, 0, stream>>>(simh, cl);
  long tot = (long)B_ * (N_ + 1) * (N_ + 1);
  k_assemble<<<(int)((tot + 255) / 256), 256, 0, stream>>>(simh, rl, cl, z0, z1,
                                                           (float*)d_out);
}
